// LPKT_33303176413361
// MI455X (gfx1250) — matrix operand order repaired and checked by reading, not compiled, not run on hardware
//
#include <hip/hip_runtime.h>

// LPKT forward for MI455X (gfx1250, wave32).
// One workgroup per batch; recurrent state entirely in the 320KB WGP LDS
// (f16 double buffer feeding WMMA + f32 master state). Main per-step GEMM is
// computed transposed, D = W4_h^T(resident VGPR A-frags) x h^T(LDS B-frags),
// so each lane owns one concept row and 8 contiguous k's -> fully vectorized
// ds_load/store_b128 epilogue and a shfl-xor h_tilde reduction.
// All non-recurrent vec-mat work (zcat prefix @ W2/W3, it @ W4_it,
// e_next @ W5_e, all_learning) is precomputed by parallel kernels.

#define Bn    32
#define Sn    128
#define NUMC  256
#define Kd    128
#define DCr   50
#define H16S  136    // halves per h16 row (row = concept c; padded for banks)
#define HF32S 132    // floats per f32 h row (padded for banks)

typedef __attribute__((ext_vector_type(16))) _Float16 v16h;
typedef __attribute__((ext_vector_type(8)))  _Float16 v8h;
typedef __attribute__((ext_vector_type(8)))  float    v8f;
typedef __attribute__((ext_vector_type(4)))  float    v4f;

#define SMEM_H16_BYTES  (2 * NUMC * H16S * 2)          // 139,264
#define SMEM_HF32_BYTES (NUMC * HF32S * 4)             // 135,168
// float scratch offsets (floats) after the h buffers
enum { F_QQ = 0,            // 512: interleaved (q_e, q_next) pairs
       F_LG = 512, F_SV = 640, F_HT = 768, F_HTN = 896,
       F_PA = 1024, F_PB = 1280, F_YP = 1536, F_TOTAL = 1792 };
#define SMEM_BYTES (SMEM_H16_BYTES + SMEM_HF32_BYTES + F_TOTAL * 4)   // ~281.6 KB < 320 KB

__device__ __forceinline__ float fast_rcp(float x) { return __builtin_amdgcn_rcpf(x); }
__device__ __forceinline__ float sigmoidf_(float x) { return fast_rcp(1.0f + __expf(-x)); }
__device__ __forceinline__ float tanhf_(float x) { return 1.0f - 2.0f * fast_rcp(1.0f + __expf(2.0f * x)); }

// ---- pre-kernel 1: allL = [e|at|a]@W1+b1 ; yin = e@W5_e + b5 ; itW4 = it@W4_it + b4 ----
__global__ __launch_bounds__(128)
void lpkt_pre1(const int* __restrict__ e_data, const int* __restrict__ at_data,
               const int* __restrict__ it_data, const float* __restrict__ a_data,
               const float* __restrict__ e_E, const float* __restrict__ at_E,
               const float* __restrict__ it_E,
               const float* __restrict__ W1, const float* __restrict__ b1,
               const float* __restrict__ W4, const float* __restrict__ b4,
               const float* __restrict__ W5, const float* __restrict__ b5,
               float* __restrict__ allL, float* __restrict__ yin, float* __restrict__ itW4)
{
    const int r = blockIdx.x;          // row in [0, B*S)
    const int k = threadIdx.x;         // 0..127
    const float a = a_data[r];
    const float* er  = e_E  + (size_t)e_data[r]  * Kd;
    const float* atr = at_E + (size_t)at_data[r] * Kd;
    const float* itr = it_E + (size_t)it_data[r] * Kd;
    float acc = b1[k], y = b5[k], s = b4[k];
#pragma unroll 4
    for (int j = 0; j < Kd; ++j) {
        const float ej = er[j];
        acc = fmaf(ej,  W1[j * Kd + k], acc);
        y   = fmaf(ej,  W5[j * Kd + k], y);
    }
#pragma unroll 4
    for (int j = 0; j < Kd; ++j) {
        acc = fmaf(atr[j], W1[(Kd + j) * Kd + k], acc);
        s   = fmaf(itr[j], W4[(2 * Kd + j) * Kd + k], s);
    }
    float wsum = 0.f;
#pragma unroll 2
    for (int j = 0; j < DCr; ++j) wsum += W1[(2 * Kd + j) * Kd + k];
    acc = fmaf(a, wsum, acc);
    allL[(size_t)r * Kd + k] = acc;
    yin [(size_t)r * Kd + k] = y;
    itW4[(size_t)r * Kd + k] = s;
}

// ---- pre-kernel 2: zpre{2,3} = [learning_pre|it|learning] @ W{2,3}[0:384] + b{2,3} ----
__global__ __launch_bounds__(128)
void lpkt_pre2(const int* __restrict__ it_data, const float* __restrict__ it_E,
               const float* __restrict__ allL,
               const float* __restrict__ W2, const float* __restrict__ b2,
               const float* __restrict__ W3, const float* __restrict__ b3,
               float* __restrict__ zpre2, float* __restrict__ zpre3)
{
    const int r = blockIdx.x;
    const int k = threadIdx.x;
    const int t = r & (Sn - 1);
    const float* itr = it_E + (size_t)it_data[r] * Kd;
    const float* lr  = allL + (size_t)r * Kd;
    float p2 = b2[k], p3 = b3[k];
    if (t != 0) {
        const float* lp = allL + (size_t)(r - 1) * Kd;
#pragma unroll 4
        for (int j = 0; j < Kd; ++j) {
            const float v = lp[j];
            p2 = fmaf(v, W2[j * Kd + k], p2);
            p3 = fmaf(v, W3[j * Kd + k], p3);
        }
    }
#pragma unroll 4
    for (int j = 0; j < Kd; ++j) {
        const float v = itr[j];
        p2 = fmaf(v, W2[(Kd + j) * Kd + k], p2);
        p3 = fmaf(v, W3[(Kd + j) * Kd + k], p3);
    }
#pragma unroll 4
    for (int j = 0; j < Kd; ++j) {
        const float v = lr[j];
        p2 = fmaf(v, W2[(2 * Kd + j) * Kd + k], p2);
        p3 = fmaf(v, W3[(2 * Kd + j) * Kd + k], p3);
    }
    zpre2[(size_t)r * Kd + k] = p2;
    zpre3[(size_t)r * Kd + k] = p3;
}

// ---------------- main scan kernel: one workgroup per batch ----------------
__global__ __launch_bounds__(256, 1)
void lpkt_scan(const int* __restrict__ e_data, const float* __restrict__ qm,
               const float* __restrict__ W2, const float* __restrict__ W3,
               const float* __restrict__ W4, const float* __restrict__ W5,
               const float* __restrict__ h0,
               const float* __restrict__ zpre2_a, const float* __restrict__ zpre3_a,
               const float* __restrict__ itW4_a,  const float* __restrict__ yin_a,
               float* __restrict__ pred)
{
    extern __shared__ char smem[];
    _Float16* h16    = (_Float16*)smem;                                  // [2][NUMC][H16S]
    float*    hstate = (float*)(smem + SMEM_H16_BYTES);                  // [NUMC][HF32S]
    float*    sf     = (float*)(smem + SMEM_H16_BYTES + SMEM_HF32_BYTES);

    const int b    = blockIdx.x;
    const int tid  = threadIdx.x;
    const int wave = tid >> 5, lane = tid & 31;
    const int lo16 = lane & 15, hi = lane >> 4;
    const int n0   = wave * 16;                        // wave's output-k tile
    const int kcol = n0 + lo16;

    const int*   eb  = e_data + b * Sn;
    const float* zp2 = zpre2_a + (size_t)b * Sn * Kd;
    const float* zp3 = zpre3_a + (size_t)b * Sn * Kd;
    const float* sIT = itW4_a  + (size_t)b * Sn * Kd;
    const float* yin = yin_a   + (size_t)b * Sn * Kd;

    // ---- resident A fragments = W4_h^T tile (output rows = k in [n0,n0+16)) ----
    v16h afrag[4];
#pragma unroll
    for (int kc = 0; kc < 4; ++kc) {
        const int kb = kc * 32;
        v16h f;
#pragma unroll
        for (int j = 0; j < 8; ++j) {
            f[j]     = (_Float16)W4[(kb + hi * 8 + j) * Kd + kcol];
            f[j + 8] = (_Float16)W4[(kb + 16 + hi * 8 + j) * Kd + kcol];
        }
        afrag[kc] = f;
    }

    // ---- init: h = h0, h_tilde = q0 . h0, pred[:,0] = 0 ----
    for (int i = tid; i < NUMC * Kd; i += 256) {
        const float v = h0[i];
        const int c = i >> 7, kk = i & 127;
        hstate[c * HF32S + kk] = v;
        h16[c * H16S + kk]     = (_Float16)v;
    }
    sf[F_QQ + 2 * tid] = qm[(size_t)eb[0] * NUMC + tid];
    if (tid == 0) pred[b * Sn] = 0.0f;
    __syncthreads();
    {
        const int kk = tid & 127, half = tid >> 7;
        float p = 0.f;
        for (int c = half * 128; c < half * 128 + 128; ++c)
            p = fmaf(sf[F_QQ + 2 * c], h0[c * Kd + kk], p);
        sf[F_PA + tid] = p;
    }
    __syncthreads();
    if (tid < 128) sf[F_HT + tid] = sf[F_PA + tid] + sf[F_PA + 128 + tid];
    __syncthreads();

    // ---- sequential scan over 127 steps ----
    for (int t = 0; t < Sn - 1; ++t) {
        const int cur = t & 1, nxt = cur ^ 1;
        _Float16* hcur = h16 + cur * (NUMC * H16S);
        _Float16* hnxt = h16 + nxt * (NUMC * H16S);
        const int eC = eb[t], eN = eb[t + 1];

        // (a) stage interleaved (q_e, q_next) pairs; prefetch next q row
        sf[F_QQ + 2 * tid]     = qm[(size_t)eC * NUMC + tid];
        sf[F_QQ + 2 * tid + 1] = qm[(size_t)eN * NUMC + tid];
        if (tid == 0 && t + 2 < Sn)
            __builtin_prefetch(&qm[(size_t)eb[t + 2] * NUMC], 0, 0);
        __syncthreads();

        // (b) recurrent slice of zcat @ W2 / W3 (h_tilde x W[384:512]); split-K x2
        {
            const int kk = tid & 127, half = tid >> 7;
            float pa = 0.f, pb = 0.f;
            const int j0 = half * 64;
#pragma unroll 4
            for (int j = j0; j < j0 + 64; ++j) {
                const float v = sf[F_HT + j];
                pa = fmaf(v, W2[(3 * Kd + j) * Kd + kk], pa);
                pb = fmaf(v, W3[(3 * Kd + j) * Kd + kk], pb);
            }
            sf[F_PA + tid] = pa;
            sf[F_PB + tid] = pb;
        }
        __syncthreads();
        if (tid < 128) {
            const float lg = tanhf_(sf[F_PA + tid] + sf[F_PA + 128 + tid] + zp2[t * Kd + tid]);
            const float gl = sigmoidf_(sf[F_PB + tid] + sf[F_PB + 128 + tid] + zp3[t * Kd + tid]);
            sf[F_LG + tid] = gl * (lg + 1.0f) * 0.5f;    // LG
        }
        __syncthreads();

        // (c) s[k] = LG @ W4_LG + (it @ W4_it + b4)[precomputed]
        {
            const int kk = tid & 127, half = tid >> 7;
            float p = 0.f;
            const int j0 = half * 64;
#pragma unroll 4
            for (int j = j0; j < j0 + 64; ++j)
                p = fmaf(sf[F_LG + j], W4[(Kd + j) * Kd + kk], p);
            sf[F_PA + tid] = p;
        }
        __syncthreads();
        if (tid < 128) sf[F_SV + tid] = sf[F_PA + tid] + sf[F_PA + 128 + tid] + sIT[t * Kd + tid];
        __syncthreads();

        // (d) D = W4_h^T x h^T on WMMA; fused state update + h_tilde shfl-reduction
        {
            const v4f svlo = *(const v4f*)&sf[F_SV + n0 + hi * 8];
            const v4f svhi = *(const v4f*)&sf[F_SV + n0 + hi * 8 + 4];
            const v4f lglo = *(const v4f*)&sf[F_LG + n0 + hi * 8];
            const v4f lghi = *(const v4f*)&sf[F_LG + n0 + hi * 8 + 4];
            float htp[8] = {0.f, 0.f, 0.f, 0.f, 0.f, 0.f, 0.f, 0.f};
            for (int ct = 0; ct < NUMC; ct += 16) {
                const int c = ct + lo16;
                const _Float16* brow = hcur + c * H16S + hi * 8;    // B: K halves 8h..8h+7 and 16+8h..16+8h+7
                v8f acc = {};
#pragma unroll
                for (int kc = 0; kc < 4; ++kc) {
                    const v8h b0 = *(const v8h*)(brow + kc * 32);
                    const v8h b1 = *(const v8h*)(brow + kc * 32 + 16);
                    const v16h bfrag = __builtin_shufflevector(
                        b0, b1, 0,1,2,3,4,5,6,7,8,9,10,11,12,13,14,15);
                    acc = __builtin_amdgcn_wmma_f32_16x16x32_f16(
                        false, afrag[kc], false, bfrag, (short)0, acc, false, false);
                }
                const float2 qq = *(const float2*)&sf[F_QQ + 2 * c]; // (q_e, q_next)
                float* hrow = &hstate[c * HF32S + n0 + hi * 8];
                const v4f hv0 = *(const v4f*)hrow;
                const v4f hv1 = *(const v4f*)(hrow + 4);
                v4f hn0, hn1;
#pragma unroll
                for (int r = 0; r < 4; ++r) {
                    hn0[r] = fmaf(sigmoidf_(acc[r]     + svlo[r]), hv0[r], qq.x * lglo[r]);
                    hn1[r] = fmaf(sigmoidf_(acc[4 + r] + svhi[r]), hv1[r], qq.x * lghi[r]);
                }
                *(v4f*)hrow       = hn0;
                *(v4f*)(hrow + 4) = hn1;
                v8h hh;
#pragma unroll
                for (int r = 0; r < 4; ++r) { hh[r] = (_Float16)hn0[r]; hh[4 + r] = (_Float16)hn1[r]; }
                *(v8h*)&hnxt[c * H16S + n0 + hi * 8] = hh;
#pragma unroll
                for (int r = 0; r < 4; ++r) {
                    htp[r]     = fmaf(qq.y, hn0[r], htp[r]);
                    htp[4 + r] = fmaf(qq.y, hn1[r], htp[4 + r]);
                }
            }
            // reduce over the 16 lanes of each half (c-subsets) -> h_tilde_new[k]
#pragma unroll
            for (int m = 1; m < 16; m <<= 1)
#pragma unroll
                for (int r = 0; r < 8; ++r) htp[r] += __shfl_xor(htp[r], m, 32);
            if (lo16 == 0) {
                const v4f h0v = {htp[0], htp[1], htp[2], htp[3]};
                const v4f h1v = {htp[4], htp[5], htp[6], htp[7]};
                *(v4f*)&sf[F_HTN + n0 + hi * 8]     = h0v;
                *(v4f*)&sf[F_HTN + n0 + hi * 8 + 4] = h1v;
            }
        }
        __syncthreads();

        // (e) y = sigmoid(h_tilde_new @ W5_h + (e_next @ W5_e + b5)[precomputed]).sum / K
        {
            const int kk = tid & 127, half = tid >> 7;
            float p = 0.f;
            const int j0 = half * 64;
#pragma unroll 4
            for (int j = j0; j < j0 + 64; ++j)
                p = fmaf(sf[F_HTN + j], W5[(Kd + j) * Kd + kk], p);
            sf[F_YP + tid] = p;
            if (tid < 128) sf[F_HT + tid] = sf[F_HTN + tid];   // carry h_tilde
        }
        __syncthreads();
        if (tid < 128)
            sf[F_YP + tid] = sigmoidf_(sf[F_YP + tid] + sf[F_YP + 128 + tid] + yin[(t + 1) * Kd + tid]);
        __syncthreads();
        if (wave == 0) {
            float v = sf[F_YP + lane] + sf[F_YP + lane + 32] +
                      sf[F_YP + lane + 64] + sf[F_YP + lane + 96];
#pragma unroll
            for (int off = 16; off > 0; off >>= 1) v += __shfl_down(v, off, 32);
            if (lane == 0) pred[b * Sn + t + 1] = v * (1.0f / Kd);
        }
        __syncthreads();
    }
}

extern "C" void kernel_launch(void* const* d_in, const int* in_sizes, int n_in,
                              void* d_out, int out_size, void* d_ws, size_t ws_size,
                              hipStream_t stream)
{
    const int*   e_data  = (const int*)  d_in[0];
    const int*   at_data = (const int*)  d_in[1];
    const int*   it_data = (const int*)  d_in[2];
    const float* a_data  = (const float*)d_in[3];
    const float* qm      = (const float*)d_in[4];
    const float* e_E     = (const float*)d_in[5];
    const float* at_E    = (const float*)d_in[6];
    const float* it_E    = (const float*)d_in[7];
    const float* W1 = (const float*)d_in[8];  const float* b1 = (const float*)d_in[9];
    const float* W2 = (const float*)d_in[10]; const float* b2 = (const float*)d_in[11];
    const float* W3 = (const float*)d_in[12]; const float* b3 = (const float*)d_in[13];
    const float* W4 = (const float*)d_in[14]; const float* b4 = (const float*)d_in[15];
    const float* W5 = (const float*)d_in[16]; const float* b5 = (const float*)d_in[17];
    const float* h0 = (const float*)d_in[18];

    float* pred = (float*)d_out;
    const size_t rows = (size_t)Bn * Sn * Kd;            // 512K floats = 2 MB each
    float* allL  = (float*)d_ws;
    float* yin   = allL  + rows;
    float* itW4  = yin   + rows;
    float* zpre2 = itW4  + rows;
    float* zpre3 = zpre2 + rows;

    lpkt_pre1<<<Bn * Sn, 128, 0, stream>>>(e_data, at_data, it_data, a_data,
                                           e_E, at_E, it_E, W1, b1, W4, b4, W5, b5,
                                           allL, yin, itW4);
    lpkt_pre2<<<Bn * Sn, 128, 0, stream>>>(it_data, it_E, allL, W2, b2, W3, b3,
                                           zpre2, zpre3);
    lpkt_scan<<<Bn, 256, SMEM_BYTES, stream>>>(e_data, qm, W2, W3, W4, W5, h0,
                                               zpre2, zpre3, itW4, yin, pred);
}
